// MultiHeadedAttention_39041252720681
// MI455X (gfx1250) — matrix operand order repaired and checked by reading, not compiled, not run on hardware
//
#include <hip/hip_runtime.h>

typedef __bf16 bf16;
typedef bf16 v16bf __attribute__((ext_vector_type(16)));
typedef bf16 v8bf  __attribute__((ext_vector_type(8)));
typedef bf16 v4bf  __attribute__((ext_vector_type(4)));
typedef float v8f  __attribute__((ext_vector_type(8)));
typedef unsigned int u32x4 __attribute__((ext_vector_type(4)));
typedef int i32x8 __attribute__((ext_vector_type(8)));
typedef int i32x4 __attribute__((ext_vector_type(4)));

#define B_SZ   4
#define T_SEQ  2048
#define NH     16
#define DH     64
#define CDIM   1024
#define C3     3072
#define MROWS  (B_SZ * T_SEQ)   // 8192

// Tensor Data Mover availability (this toolchain: 6-arg builtin)
#if defined(__has_builtin)
#if __has_builtin(__builtin_amdgcn_tensor_load_to_lds) && \
    __has_builtin(__builtin_amdgcn_s_wait_tensorcnt)
#define HAVE_TDM 1
#endif
#endif
#ifndef HAVE_TDM
#define HAVE_TDM 0
#endif

static __device__ inline v8f vzero8() {
    v8f z;
#pragma unroll
    for (int i = 0; i < 8; ++i) z[i] = 0.0f;
    return z;
}

static __device__ inline v16bf cat8(v8bf lo, v8bf hi) {
    v16bf r;
#pragma unroll
    for (int i = 0; i < 8; ++i) { r[i] = lo[i]; r[i + 8] = hi[i]; }
    return r;
}

static __device__ inline v8f mma32(v16bf a, v16bf b, v8f c) {
    return __builtin_amdgcn_wmma_f32_16x16x32_bf16(
        /*neg_a=*/false, a, /*neg_b=*/false, b,
        /*c_mod=*/(short)0, c, /*reuse_a=*/false, /*reuse_b=*/false);
}

#if HAVE_TDM
// LDS byte offset of a __shared__ object: generic (flat) address of LDS
// aperture keeps the LDS offset in addr[31:0].
static __device__ inline unsigned lds_addr_of(const void* p) {
    return (unsigned)(unsigned long long)p;
}

// Issue a 2D TDM load: tile dim0 x dim1 of bf16 (dim0 contiguous in memory,
// row stride = stride_elems). g1flags carries optional LDS padding bits.
// D# bitfields per CDNA5 ISA (groups 0/1; groups 2/3 and
// the trailing group zeroed => 2D tensor, no iteration/gather).
static __device__ __attribute__((always_inline)) void tdm_load_2d(
    unsigned lds_byte, const bf16* gptr, unsigned dim0, unsigned dim1,
    unsigned long long stride_elems, unsigned g1flags) {
    unsigned long long ga = (unsigned long long)gptr;
    u32x4 g0;
    g0[0] = 1u;                                              // count=1, user mode
    g0[1] = lds_byte;                                        // lds_addr[31:0]
    g0[2] = (unsigned)ga;                                    // global_addr[31:0]
    g0[3] = (unsigned)((ga >> 32) & 0x01FFFFFFu) | (2u << 30); // ga[56:32], type=2
    i32x8 g1;
    g1[0] = (int)((1u << 16) | g1flags);                     // data_size=2B (+pad)
    g1[1] = (int)((dim0 & 0xFFFFu) << 16);                   // tensor_dim0[15:0]
    g1[2] = (int)((dim0 >> 16) | ((dim1 & 0xFFFFu) << 16));  // td0[31:16], td1[15:0]
    g1[3] = (int)((dim1 >> 16) | ((dim0 & 0xFFFFu) << 16));  // td1[31:16], tile_dim0
    g1[4] = (int)(dim1 & 0xFFFFu);                           // tile_dim1, tile_dim2=0
    g1[5] = (int)(stride_elems & 0xFFFFFFFFull);             // dim0_stride[31:0]
    g1[6] = (int)((stride_elems >> 32) & 0xFFFFull);         // dim0_stride[47:32]
    g1[7] = 0;
    i32x4 z4;
    z4[0] = 0; z4[1] = 0; z4[2] = 0; z4[3] = 0;
    i32x8 z8;
#pragma unroll
    for (int i = 0; i < 8; ++i) z8[i] = 0;
    __builtin_amdgcn_tensor_load_to_lds(g0, g1, z4, z4, z8, 0);
}
#endif

// ---------------------------------------------------------------------------
// fp32 -> bf16 elementwise convert (n divisible by 4)
// ---------------------------------------------------------------------------
__global__ __launch_bounds__(256) void cvt_f32_bf16(
    const float* __restrict__ in, bf16* __restrict__ out, int n) {
    int i = (blockIdx.x * 256 + threadIdx.x) * 4;
    if (i < n) {
        float4 v = *(const float4*)(in + i);
        v4bf o;
        o[0] = (bf16)v.x; o[1] = (bf16)v.y; o[2] = (bf16)v.z; o[3] = (bf16)v.w;
        *(v4bf*)(out + i) = o;
    }
}

// ---------------------------------------------------------------------------
// fp32 [K,N] -> bf16 [N,K] transpose+convert
// ---------------------------------------------------------------------------
__global__ __launch_bounds__(256) void transpose_cvt(
    const float* __restrict__ w, bf16* __restrict__ wT, int K, int N) {
    int idx = blockIdx.x * 256 + threadIdx.x;
    if (idx < K * N) {
        int k = idx / N, n = idx - k * N;
        wT[(size_t)n * K + k] = (bf16)w[idx];
    }
}

// ---------------------------------------------------------------------------
// GEMM: out[m][n] = sum_k A[m][k] * Bt[n][k] + bias[n]
// A: bf16 [M,K], Bt: bf16 [N,K] (K contiguous in both).
// Block tile 128x128, BK=32, 256 threads = 8 waves (wave32), each wave 64x32.
// TDM path: double-buffered LDS tiles, wave 0 drives the Tensor Data Mover
// (next tile DMA overlaps WMMA on current tile, tracked by TENSORcnt).
// ---------------------------------------------------------------------------
#define LDSTR 48  // bf16 stride per row: 32 data + 16 pad (keeps 16B align)

template <bool OUT_BF16>
__global__ __launch_bounds__(256) void gemm_tn(
    const bf16* __restrict__ A, const bf16* __restrict__ Bt,
    const float* __restrict__ bias, void* __restrict__ outp,
    int M, int N, int K) {
#if HAVE_TDM
    __shared__ bf16 sA[2 * 128 * LDSTR];
    __shared__ bf16 sB[2 * 128 * LDSTR];
#else
    __shared__ bf16 sA[128 * LDSTR];
    __shared__ bf16 sB[128 * LDSTR];
#endif

    const int tid  = threadIdx.x;
    const int wave = tid >> 5;
    const int lane = tid & 31;
    const int hl   = lane >> 4;   // half-group 0/1
    const int ln   = lane & 15;
    const int wm   = wave & 1;    // 2 waves along M
    const int wn   = wave >> 1;   // 4 waves along N
    const int m0   = blockIdx.y * 128;
    const int n0   = blockIdx.x * 128;

    v8f acc[4][2];
#pragma unroll
    for (int i = 0; i < 4; ++i)
#pragma unroll
        for (int j = 0; j < 2; ++j) acc[i][j] = vzero8();

#if HAVE_TDM
    // LDS row padding: rows of 32 bf16 = 16 DWORDs -> pad_interval code 3
    // (pad every 16 DWORDs), pad_amount code 7 (8 DWORDs = 16 bf16) => stride 48.
    const unsigned PADF  = (1u << 20) | (3u << 22) | (7u << 25);
    const unsigned ldsA0 = lds_addr_of(sA);
    const unsigned ldsB0 = lds_addr_of(sB);
    const unsigned BUFB  = 128u * LDSTR * 2u;  // bytes per buffer

    if (wave == 0) {
        tdm_load_2d(ldsA0, A + (size_t)m0 * K, 32, 128, (unsigned long long)K, PADF);
        tdm_load_2d(ldsB0, Bt + (size_t)n0 * K, 32, 128, (unsigned long long)K, PADF);
    }
    int cur = 0;
    const int NK = K / 32;
    for (int t = 0; t < NK; ++t) {
        if (wave == 0) __builtin_amdgcn_s_wait_tensorcnt(0);
        __syncthreads();  // publish buf[cur] to all waves
        if (wave == 0 && t + 1 < NK) {
            int k0n = (t + 1) * 32;
            tdm_load_2d(ldsA0 + (unsigned)(cur ^ 1) * BUFB,
                        A + (size_t)m0 * K + k0n, 32, 128,
                        (unsigned long long)K, PADF);
            tdm_load_2d(ldsB0 + (unsigned)(cur ^ 1) * BUFB,
                        Bt + (size_t)n0 * K + k0n, 32, 128,
                        (unsigned long long)K, PADF);
        }
        const bf16* bA = sA + cur * 128 * LDSTR;
        const bf16* bB = sB + cur * 128 * LDSTR;
#else
    for (int k0 = 0; k0 < K; k0 += 32) {
        // Stage 128x32 bf16 of A and Bt: 512 chunks of 8 bf16 each.
#pragma unroll
        for (int c = tid; c < 512; c += 256) {
            int row = c >> 2, seg = c & 3;
            *(v8bf*)(&sA[row * LDSTR + seg * 8]) =
                *(const v8bf*)(A + (size_t)(m0 + row) * K + k0 + seg * 8);
            *(v8bf*)(&sB[row * LDSTR + seg * 8]) =
                *(const v8bf*)(Bt + (size_t)(n0 + row) * K + k0 + seg * 8);
        }
        __syncthreads();
        const bf16* bA = sA;
        const bf16* bB = sB;
#endif
        v16bf af[4], bfr[2];
#pragma unroll
        for (int i = 0; i < 4; ++i) {
            int m = wm * 64 + i * 16 + ln;
            v8bf lo = *(const v8bf*)(&bA[m * LDSTR + hl * 8]);
            v8bf hi = *(const v8bf*)(&bA[m * LDSTR + 16 + hl * 8]);
            af[i] = cat8(lo, hi);
        }
#pragma unroll
        for (int j = 0; j < 2; ++j) {
            int n = wn * 32 + j * 16 + ln;
            v8bf lo = *(const v8bf*)(&bB[n * LDSTR + hl * 8]);
            v8bf hi = *(const v8bf*)(&bB[n * LDSTR + 16 + hl * 8]);
            bfr[j] = cat8(lo, hi);
        }
#pragma unroll
        for (int i = 0; i < 4; ++i)
#pragma unroll
            for (int j = 0; j < 2; ++j)
                acc[i][j] = mma32(af[i], bfr[j], acc[i][j]);
        __syncthreads();
#if HAVE_TDM
        cur ^= 1;
#endif
    }

    // Epilogue: D layout m = r + 8*hl, n = ln
#pragma unroll
    for (int i = 0; i < 4; ++i)
#pragma unroll
        for (int j = 0; j < 2; ++j) {
            int gn = n0 + wn * 32 + j * 16 + ln;
            float bv = bias[gn];
#pragma unroll
            for (int r = 0; r < 8; ++r) {
                int gm = m0 + wm * 64 + i * 16 + r + 8 * hl;
                float v = acc[i][j][r] + bv;
                if (OUT_BF16)
                    ((bf16*)outp)[(size_t)gm * N + gn] = (bf16)v;
                else
                    ((float*)outp)[(size_t)gm * N + gn] = v;
            }
        }
}

// ---------------------------------------------------------------------------
// Fused causal flash attention.
// qkv: bf16 [B*T, 3C] (Q at col 0, K at col C, V at col 2C; head h at h*DH).
// out: bf16 [B*T, C].
// Grid: (T/64, H, B), 128 threads = 4 waves; wave w owns q rows [qb+16w, +16).
// K tile fetched by TDM (overlapped with the manual V-transpose staging).
// ---------------------------------------------------------------------------
__global__ __launch_bounds__(128) void attn_fused(
    const bf16* __restrict__ qkv, bf16* __restrict__ out) {
    __shared__ bf16 sK[32 * 64];        // [kv][dd], dd contiguous
    __shared__ bf16 sVt[64 * LDSTR];    // [dd][kv], kv contiguous (padded)
    __shared__ bf16 sP[4][16 * LDSTR];  // per-wave P tile [m][kv]

    const int qblk = blockIdx.x;
    const int h    = blockIdx.y;
    const int b    = blockIdx.z;
    const int tid  = threadIdx.x;
    const int wave = tid >> 5;
    const int lane = tid & 31;
    const int hl   = lane >> 4;
    const int ln   = lane & 15;

    const int qbase_blk = qblk * 64;
    const int qbase_w   = qbase_blk + wave * 16;

    // Q fragments for both 32-wide d slices (A layout: m = ln, k via hl/e).
    const bf16* qrow = qkv + (size_t)(b * T_SEQ + qbase_w + ln) * C3 + h * DH;
    v16bf qf[2];
#pragma unroll
    for (int s2 = 0; s2 < 2; ++s2) {
        int s = s2 * 32;
        v8bf lo = *(const v8bf*)(qrow + s + hl * 8);
        v8bf hi = *(const v8bf*)(qrow + s + 16 + hl * 8);
        qf[s2] = cat8(lo, hi);
    }

    v8f o[4];
#pragma unroll
    for (int j = 0; j < 4; ++j) o[j] = vzero8();
    float mrow[8], lrow[8];
#pragma unroll
    for (int r = 0; r < 8; ++r) { mrow[r] = -1e30f; lrow[r] = 0.0f; }

#if HAVE_TDM
    const unsigned ldsK = lds_addr_of(sK);
#endif

    const int kvend = qbase_blk + 64;
    for (int kv0 = 0; kv0 < kvend; kv0 += 32) {
        // ---- stage K tile (32 x 64 bf16, row-major) ----
#if HAVE_TDM
        if (wave == 0)
            tdm_load_2d(ldsK,
                        qkv + (size_t)(b * T_SEQ + kv0) * C3 + CDIM + h * DH,
                        64, 32, (unsigned long long)C3, 0u);
#else
#pragma unroll
        for (int c = tid; c < 256; c += 128) {
            int kv = c >> 3, seg = c & 7;
            *(v8bf*)(&sK[kv * 64 + seg * 8]) = *(const v8bf*)(
                qkv + (size_t)(b * T_SEQ + kv0 + kv) * C3 + CDIM + h * DH + seg * 8);
        }
#endif
        // ---- stage V tile transposed (64 dd x 32 kv): b128 loads, b16 scatter ----
#pragma unroll
        for (int c = tid; c < 256; c += 128) {
            int kv = c >> 3, seg = c & 7;
            v8bf ch = *(const v8bf*)(
                qkv + (size_t)(b * T_SEQ + kv0 + kv) * C3 + 2 * CDIM + h * DH + seg * 8);
#pragma unroll
            for (int e = 0; e < 8; ++e)
                sVt[(seg * 8 + e) * LDSTR + kv] = ch[e];
        }
#if HAVE_TDM
        if (wave == 0) __builtin_amdgcn_s_wait_tensorcnt(0);
#endif
        __syncthreads();

        // ---- S = Q * K^T  (two 16x16 tiles over kv 0..15 / 16..31) ----
        v8f s0 = vzero8(), s1 = vzero8();
#pragma unroll
        for (int s2 = 0; s2 < 2; ++s2) {
            int s = s2 * 32;
            v8bf a0 = *(const v8bf*)(&sK[ln * 64 + s + hl * 8]);
            v8bf a1 = *(const v8bf*)(&sK[ln * 64 + s + 16 + hl * 8]);
            v8bf b0 = *(const v8bf*)(&sK[(16 + ln) * 64 + s + hl * 8]);
            v8bf b1 = *(const v8bf*)(&sK[(16 + ln) * 64 + s + 16 + hl * 8]);
            s0 = mma32(qf[s2], cat8(a0, a1), s0);
            s1 = mma32(qf[s2], cat8(b0, b1), s1);
        }

        // ---- online softmax (row r maps to q_t = qbase_w + r + 8*hl) ----
        const bool need_mask = (kv0 + 31 > qbase_w);
#pragma unroll
        for (int r = 0; r < 8; ++r) {
            int qt = qbase_w + r + 8 * hl;
            float v0 = s0[r] * 0.125f;   // 1/sqrt(64)
            float v1 = s1[r] * 0.125f;
            if (need_mask) {
                if (kv0 + ln > qt)      v0 = -1e30f;
                if (kv0 + 16 + ln > qt) v1 = -1e30f;
            }
            float mc = fmaxf(v0, v1);
#pragma unroll
            for (int off = 8; off; off >>= 1)
                mc = fmaxf(mc, __shfl_xor(mc, off, 32));
            float mnew  = fmaxf(mrow[r], mc);
            float scale = __expf(mrow[r] - mnew);
            float p0 = __expf(v0 - mnew);
            float p1 = __expf(v1 - mnew);
            float ps = p0 + p1;
#pragma unroll
            for (int off = 8; off; off >>= 1)
                ps += __shfl_xor(ps, off, 32);
            lrow[r] = lrow[r] * scale + ps;
            mrow[r] = mnew;
            o[0][r] *= scale; o[1][r] *= scale;
            o[2][r] *= scale; o[3][r] *= scale;
            // Re-stripe P through per-wave LDS (D layout -> A layout)
            sP[wave][(r + 8 * hl) * LDSTR + ln]      = (bf16)p0;
            sP[wave][(r + 8 * hl) * LDSTR + 16 + ln] = (bf16)p1;
        }

        // ---- O += P * V ----
        v8bf plo = *(const v8bf*)(&sP[wave][ln * LDSTR + hl * 8]);
        v8bf phi = *(const v8bf*)(&sP[wave][ln * LDSTR + 16 + hl * 8]);
        v16bf pf = cat8(plo, phi);
#pragma unroll
        for (int j = 0; j < 4; ++j) {
            v8bf vlo = *(const v8bf*)(&sVt[(j * 16 + ln) * LDSTR + hl * 8]);
            v8bf vhi = *(const v8bf*)(&sVt[(j * 16 + ln) * LDSTR + 16 + hl * 8]);
            o[j] = mma32(pf, cat8(vlo, vhi), o[j]);
        }
        __syncthreads();
    }

    // ---- epilogue: normalize and emit bf16 [B*T, C] ----
#pragma unroll
    for (int r = 0; r < 8; ++r) {
        float inv = 1.0f / lrow[r];
        size_t row = (size_t)(b * T_SEQ + qbase_w + r + 8 * hl);
#pragma unroll
        for (int j = 0; j < 4; ++j)
            out[row * CDIM + h * DH + j * 16 + ln] = (bf16)(o[j][r] * inv);
    }
}

// ---------------------------------------------------------------------------
// Host-side orchestration
// ---------------------------------------------------------------------------
extern "C" void kernel_launch(void* const* d_in, const int* in_sizes, int n_in,
                              void* d_out, int out_size, void* d_ws, size_t ws_size,
                              hipStream_t stream) {
    (void)in_sizes; (void)n_in; (void)out_size; (void)ws_size;
    const float* x      = (const float*)d_in[0];   // [B,T,C]
    const float* w_qkv  = (const float*)d_in[1];   // [C,3C]
    const float* b_qkv  = (const float*)d_in[2];   // [3C]
    const float* w_proj = (const float*)d_in[3];   // [C,C]
    const float* b_proj = (const float*)d_in[4];   // [C]

    char* ws = (char*)d_ws;
    const size_t XB_BYTES    = (size_t)MROWS * CDIM * 2;   // 16 MB
    const size_t WQKVT_BYTES = (size_t)C3 * CDIM * 2;      //  6 MB
    const size_t WPROJ_BYTES = (size_t)CDIM * CDIM * 2;    //  2 MB
    const size_t QKVB_BYTES  = (size_t)MROWS * C3 * 2;     // 48 MB

    bf16* xb     = (bf16*)(ws);
    bf16* wqkvT  = (bf16*)(ws + XB_BYTES);
    bf16* wprojT = (bf16*)(ws + XB_BYTES + WQKVT_BYTES);
    bf16* qkvb   = (bf16*)(ws + XB_BYTES + WQKVT_BYTES + WPROJ_BYTES);
    bf16* attnb  = (bf16*)(ws + XB_BYTES + WQKVT_BYTES + WPROJ_BYTES + QKVB_BYTES);

    // 1. Convert x to bf16
    {
        int n = MROWS * CDIM;
        cvt_f32_bf16<<<n / (256 * 4), 256, 0, stream>>>(x, xb, n);
    }
    // 2. Transposed bf16 weights
    transpose_cvt<<<(CDIM * C3) / 256, 256, 0, stream>>>(w_qkv, wqkvT, CDIM, C3);
    transpose_cvt<<<(CDIM * CDIM) / 256, 256, 0, stream>>>(w_proj, wprojT, CDIM, CDIM);
    // 3. QKV projection (bf16 out, bias fused)
    gemm_tn<true><<<dim3(C3 / 128, MROWS / 128), 256, 0, stream>>>(
        xb, wqkvT, b_qkv, (void*)qkvb, MROWS, C3, CDIM);
    // 4. Fused causal attention
    attn_fused<<<dim3(T_SEQ / 64, NH, B_SZ), 128, 0, stream>>>(qkvb, attnb);
    // 5. Output projection (fp32 out, bias fused)
    gemm_tn<false><<<dim3(CDIM / 128, MROWS / 128), 256, 0, stream>>>(
        attnb, wprojT, b_proj, d_out, MROWS, CDIM, CDIM);
}
